// MultiHeadSelfAttention_59347858096812
// MI455X (gfx1250) — compile-verified, matrix operand order checked by reading, not run on hardware
//
#include <hip/hip_runtime.h>
#include <hip/hip_bf16.h>

typedef __attribute__((ext_vector_type(16))) __bf16 v16bf;
typedef __attribute__((ext_vector_type(8)))  __bf16 v8bf;
typedef __attribute__((ext_vector_type(8)))  float  v8f;
typedef int v4i_gcc __attribute__((vector_size(16)));

#define S_LEN  2048
#define DMODEL 512
#define NH     8
#define HD     64
#define BATCH  4
#define SPAN   128
#define NEG_INF_F (-1e9f)

#if defined(__has_builtin)
#if __has_builtin(__builtin_amdgcn_global_load_async_to_lds_b128)
#define HAVE_ASYNC_LDS 1
#endif
#endif

// Copy 16 bytes global -> LDS. Async-to-LDS (ASYNCcnt) when available,
// else a plain load+store through VGPRs.
static __device__ inline void cp16_async(__bf16* l, const __bf16* g) {
#ifdef HAVE_ASYNC_LDS
  __builtin_amdgcn_global_load_async_to_lds_b128(
      (__attribute__((address_space(1))) v4i_gcc*)(unsigned long long)g,
      (__attribute__((address_space(3))) v4i_gcc*)(unsigned)(unsigned long long)l,
      0, 0);
#else
  *(v8bf*)l = *(const v8bf*)g;
#endif
}

// Wait until ASYNCcnt <= N (async loads complete in order, so N = in-flight
// depth of the *next* prefetch stage guarantees the current stage landed).
template <int N>
static __device__ inline void async_wait_le() {
#ifdef HAVE_ASYNC_LDS
#if __has_builtin(__builtin_amdgcn_s_wait_asynccnt)
  __builtin_amdgcn_s_wait_asynccnt(N);
#else
  asm volatile("s_wait_asynccnt %0" ::"i"(N) : "memory");
#endif
#endif
}

// Load a 16-bit A/B WMMA fragment (16x32, K innermost in memory).
// Lane L (col/row = L%16, hi = L/16) needs K halves [hi*8, hi*8+8) and
// [16+hi*8, 16+hi*8+8) -> two contiguous 16-byte loads.
static __device__ inline v16bf frag_ld(const __bf16* p, int hi) {
  union { v16bf f; v8bf h[2]; } u;
  u.h[0] = *(const v8bf*)(p + hi * 8);
  u.h[1] = *(const v8bf*)(p + 16 + hi * 8);
  return u.f;
}

static __device__ inline v8f bf16_wmma(v16bf a, v16bf b, v8f c) {
  return __builtin_amdgcn_wmma_f32_16x16x32_bf16(false, a, false, b, (short)0, c,
                                                 false, false);
}

__global__ void k_cast_x(const float* __restrict__ x, __bf16* __restrict__ xb, int n) {
  int i = blockIdx.x * 256 + threadIdx.x;
  if (i < n) xb[i] = (__bf16)x[i];
}

// wT[n*K + k] = w[k*N + n]  (cast fp32 -> bf16)
__global__ void k_transpose_cast(const float* __restrict__ w, __bf16* __restrict__ wT,
                                 int K, int N) {
  int i = blockIdx.x * 256 + threadIdx.x;
  if (i < K * N) {
    int n = i / K;
    int k = i % K;
    wT[i] = (__bf16)w[(size_t)k * N + n];
  }
}

// Core: block = 4 waves, each wave computes a 32x64 tile; block = 128 rows x 64 cols.
// B tile (shared by all 4 waves) is staged into double-buffered LDS with
// async-to-LDS prefetch one K-step ahead (2 async instrs in flight per stage).
static __device__ inline void gemm_core(const __bf16* __restrict__ A,
                                        const __bf16* __restrict__ BT,
                                        int m0w, int n0, __bf16 (*ldsB)[64][32],
                                        v8f acc[2][4]) {
  const int tid  = threadIdx.x;
  const int lane = tid & 31;
  const int col  = lane & 15;
  const int hi   = lane >> 4;
  const __bf16* ar0 = A + (size_t)(m0w + col) * DMODEL;
  const __bf16* ar1 = A + (size_t)(m0w + 16 + col) * DMODEL;
  // staging: B tile = 64 rows (n) x 32 halves (k) = 256 16B segments, 2 per thread
  const int s0 = tid * 2, s1 = s0 + 1;
  const int br0 = s0 >> 2, bc0 = (s0 & 3) * 8;
  const int br1 = s1 >> 2, bc1 = (s1 & 3) * 8;
  const __bf16* b0 = BT + (size_t)(n0 + br0) * DMODEL + bc0;
  const __bf16* b1 = BT + (size_t)(n0 + br1) * DMODEL + bc1;

  auto stage = [&](int buf, int kb) {
    cp16_async(&ldsB[buf][br0][bc0], b0 + kb);
    cp16_async(&ldsB[buf][br1][bc1], b1 + kb);
  };

  const int NSTEP = DMODEL / 32;
  stage(0, 0);
#pragma unroll 1
  for (int i = 0; i < NSTEP; ++i) {
    const int cur = i & 1;
    if (i + 1 < NSTEP) {
      stage(cur ^ 1, (i + 1) * 32);  // prefetch next tile into other buffer
      async_wait_le<2>();            // current tile done; next may be in flight
    } else {
      async_wait_le<0>();
    }
    __syncthreads();
    const int kb = i * 32;
    v16bf a0 = frag_ld(ar0 + kb, hi);
    v16bf a1 = frag_ld(ar1 + kb, hi);
#pragma unroll
    for (int t = 0; t < 4; ++t) {
      v16bf bfr = frag_ld(&ldsB[cur][t * 16 + col][0], hi);
      acc[0][t] = bf16_wmma(a0, bfr, acc[0][t]);
      acc[1][t] = bf16_wmma(a1, bfr, acc[1][t]);
    }
    __syncthreads();  // all waves done reading buf `cur` before it is re-staged
  }
}

// qkv = xb[8192,512] @ wqkv[512,1536]; scatter into q/k (B,H,S,hd) and vT (B,H,hd,S)
__global__ void __launch_bounds__(128, 1)
k_qkv_gemm(const __bf16* __restrict__ xb, const __bf16* __restrict__ wT,
           __bf16* __restrict__ q, __bf16* __restrict__ kk,
           __bf16* __restrict__ vT) {
  __shared__ __bf16 ldsB[2][64][32];
  const int lane = threadIdx.x & 31;
  const int wave = threadIdx.x >> 5;
  const int col  = lane & 15;
  const int hi   = lane >> 4;
  const int m0w = blockIdx.x * 128 + wave * 32;
  const int n0  = blockIdx.y * 64;

  v8f acc[2][4] = {};
  gemm_core(xb, wT, m0w, n0, ldsB, acc);

  const float scale = 0.044194173824159216f;  // 512^-0.5 folded into Q
#pragma unroll
  for (int sub = 0; sub < 2; ++sub) {
#pragma unroll
    for (int t = 0; t < 4; ++t) {
      int n = n0 + t * 16 + col;
      int h = n / 192, rem = n % 192;
      int tt = rem / 64, d = rem % 64;
#pragma unroll
      for (int r = 0; r < 8; ++r) {
        int m = m0w + sub * 16 + r + 8 * hi;
        int b = m >> 11, s = m & (S_LEN - 1);
        float val = acc[sub][t][r];
        if (tt == 0)
          q[(((size_t)b * NH + h) * S_LEN + s) * HD + d] = (__bf16)(val * scale);
        else if (tt == 1)
          kk[(((size_t)b * NH + h) * S_LEN + s) * HD + d] = (__bf16)val;
        else
          vT[(((size_t)b * NH + h) * HD + d) * S_LEN + s] = (__bf16)val;
      }
    }
  }
}

// out[8192,512] = attnout[8192,512] @ wproj[512,512], fp32 result
__global__ void __launch_bounds__(128, 1)
k_proj_gemm(const __bf16* __restrict__ ab, const __bf16* __restrict__ wT,
            float* __restrict__ out) {
  __shared__ __bf16 ldsB[2][64][32];
  const int lane = threadIdx.x & 31;
  const int wave = threadIdx.x >> 5;
  const int col  = lane & 15;
  const int hi   = lane >> 4;
  const int m0w = blockIdx.x * 128 + wave * 32;
  const int n0  = blockIdx.y * 64;

  v8f acc[2][4] = {};
  gemm_core(ab, wT, m0w, n0, ldsB, acc);

#pragma unroll
  for (int sub = 0; sub < 2; ++sub) {
#pragma unroll
    for (int t = 0; t < 4; ++t) {
      int n = n0 + t * 16 + col;
#pragma unroll
      for (int r = 0; r < 8; ++r) {
        int m = m0w + sub * 16 + r + 8 * hi;
        out[(size_t)m * DMODEL + n] = acc[sub][t][r];
      }
    }
  }
}

// Banded flash attention: block = 64 query rows of one (b,h); 4 waves x 16 rows.
// K/V chunks (shared by all 4 waves) staged in double-buffered LDS with
// async-to-LDS prefetch one chunk ahead (4 async instrs in flight per stage).
__global__ void __launch_bounds__(128, 1)
k_band_attn(const __bf16* __restrict__ qm, const __bf16* __restrict__ km,
            const __bf16* __restrict__ vT, __bf16* __restrict__ attnout) {
  __shared__ __bf16 ldsK[2][32][64];   // keys jb..jb+31, hd innermost
  __shared__ __bf16 ldsV[2][64][32];   // hd rows, keys innermost (from vT)
  __shared__ __bf16 plds[4][16][32];   // wave-private P tiles
  const int tid  = threadIdx.x;
  const int lane = tid & 31;
  const int wave = tid >> 5;
  const int col  = lane & 15;
  const int hi   = lane >> 4;
  const int q64 = blockIdx.x & 31;   // S/64 tiles
  const int bh  = blockIdx.x >> 5;   // B*H
  const int b   = bh >> 3;
  const int h   = bh & 7;
  const int q0  = q64 * 64;
  const int r0  = q0 + wave * 16;

  const __bf16* qbh = qm + (size_t)bh * S_LEN * HD;
  const __bf16* kbh = km + (size_t)bh * S_LEN * HD;
  const __bf16* vbh = vT + (size_t)bh * HD * S_LEN;

  // Q fragments for this wave's 16 rows (hd=64 -> two K=32 fragments)
  const __bf16* qrow = qbh + (size_t)(r0 + col) * HD;
  v16bf qa0 = frag_ld(qrow, hi);
  v16bf qa1 = frag_ld(qrow + 32, hi);

  v8f o[4] = {};
  float mrow[8], lrow[8];
#pragma unroll
  for (int r = 0; r < 8; ++r) { mrow[r] = -3.0e38f; lrow[r] = 0.f; }

  // staging split: K chunk 32x64 halves = 256 segs; V chunk 64x32 halves = 256 segs
  const int s0 = tid * 2, s1 = s0 + 1;
  const int kr0 = s0 >> 3, kc0 = (s0 & 7) * 8;   // ldsK row / half-offset
  const int kr1 = s1 >> 3, kc1 = (s1 & 7) * 8;
  const int vr0 = s0 >> 2, vc0 = (s0 & 3) * 8;   // ldsV row / half-offset
  const int vr1 = s1 >> 2, vc1 = (s1 & 3) * 8;

  auto stage_kv = [&](int buf, int jb) {
    cp16_async(&ldsK[buf][kr0][kc0], kbh + (size_t)(jb + kr0) * HD + kc0);
    cp16_async(&ldsK[buf][kr1][kc1], kbh + (size_t)(jb + kr1) * HD + kc1);
    cp16_async(&ldsV[buf][vr0][vc0], vbh + (size_t)vr0 * S_LEN + jb + vc0);
    cp16_async(&ldsV[buf][vr1][vc1], vbh + (size_t)vr1 * S_LEN + jb + vc1);
  };

  int js = q0 - SPAN; if (js < 0) js = 0; js &= ~31;
  int je = q0 + 64 + SPAN; if (je > S_LEN) je = S_LEN;

  stage_kv(0, js);
  int it = 0;
#pragma unroll 1
  for (int jb = js; jb < je; jb += 32, ++it) {
    const int cur = it & 1;
    if (jb + 32 < je) {
      stage_kv(cur ^ 1, jb + 32);  // prefetch next chunk into other buffer
      async_wait_le<4>();          // current chunk landed; next may be in flight
    } else {
      async_wait_le<0>();
    }
    __syncthreads();

    // ---- scores: two 16x16 tiles over keys [jb, jb+32) ----
    v8f st[2];
#pragma unroll
    for (int t = 0; t < 2; ++t) {
      const __bf16* krow = &ldsK[cur][t * 16 + col][0];
      v16bf kf0 = frag_ld(krow, hi);
      v16bf kf1 = frag_ld(krow + 32, hi);
      v8f c = {};
      c = bf16_wmma(qa0, kf0, c);
      c = bf16_wmma(qa1, kf1, c);
      st[t] = c;
    }
    // ---- band mask + online softmax (row lives in one 16-lane half) ----
    float p0[8], p1[8], alpha[8];
#pragma unroll
    for (int r = 0; r < 8; ++r) {
      int i = r0 + r + 8 * hi;
      int j0 = jb + col, j1 = jb + 16 + col;
      float sc0 = st[0][r], sc1 = st[1][r];
      int d0 = i - j0; if (d0 < 0) d0 = -d0;
      int d1 = i - j1; if (d1 < 0) d1 = -d1;
      sc0 = (d0 <= SPAN) ? sc0 : NEG_INF_F;
      sc1 = (d1 <= SPAN) ? sc1 : NEG_INF_F;
      float mx = fmaxf(sc0, sc1);
      mx = fmaxf(mx, __shfl_xor(mx, 1, 32));
      mx = fmaxf(mx, __shfl_xor(mx, 2, 32));
      mx = fmaxf(mx, __shfl_xor(mx, 4, 32));
      mx = fmaxf(mx, __shfl_xor(mx, 8, 32));
      float mnew = fmaxf(mrow[r], mx);
      alpha[r] = __expf(mrow[r] - mnew);
      mrow[r] = mnew;
      p0[r] = __expf(sc0 - mnew);
      p1[r] = __expf(sc1 - mnew);
      float ps = p0[r] + p1[r];
      ps += __shfl_xor(ps, 1, 32);
      ps += __shfl_xor(ps, 2, 32);
      ps += __shfl_xor(ps, 4, 32);
      ps += __shfl_xor(ps, 8, 32);
      lrow[r] = lrow[r] * alpha[r] + ps;
    }
    // ---- P: C-layout -> A-layout through wave-private LDS ----
#pragma unroll
    for (int r = 0; r < 8; ++r) {
      int M = r + 8 * hi;
      plds[wave][M][col]      = (__bf16)p0[r];
      plds[wave][M][col + 16] = (__bf16)p1[r];
    }
    asm volatile("s_wait_dscnt 0" ::: "memory");  // wave-internal DS ordering
    v16bf pa = frag_ld(&plds[wave][col][0], hi);
    // ---- rescale O, accumulate P @ V from LDS ----
#pragma unroll
    for (int t = 0; t < 4; ++t) {
#pragma unroll
      for (int r = 0; r < 8; ++r) o[t][r] *= alpha[r];
      v16bf vf = frag_ld(&ldsV[cur][t * 16 + col][0], hi);
      o[t] = bf16_wmma(pa, vf, o[t]);
    }
    __syncthreads();  // all waves done with buf `cur` before it is re-staged
  }

  // ---- normalize, write attention output [B,S,H*hd] (bf16 for final GEMM) ----
#pragma unroll
  for (int r = 0; r < 8; ++r) {
    int i = r0 + r + 8 * hi;
    float rinv = 1.0f / lrow[r];
    size_t base = ((size_t)(b * S_LEN + i)) * DMODEL + h * HD;
#pragma unroll
    for (int t = 0; t < 4; ++t)
      attnout[base + t * 16 + col] = (__bf16)(o[t][r] * rinv);
  }
}

extern "C" void kernel_launch(void* const* d_in, const int* in_sizes, int n_in,
                              void* d_out, int out_size, void* d_ws, size_t ws_size,
                              hipStream_t stream) {
  const float* x      = (const float*)d_in[0];
  const float* w_qkv  = (const float*)d_in[1];
  const float* w_proj = (const float*)d_in[2];
  float* out = (float*)d_out;

  char* ws = (char*)d_ws;
  size_t off = 0;
  const size_t BS = (size_t)BATCH * S_LEN;              // 8192
  __bf16* xb      = (__bf16*)(ws + off); off += BS * DMODEL * 2;                 // 8 MB
  __bf16* wqkvT   = (__bf16*)(ws + off); off += (size_t)3 * DMODEL * DMODEL * 2; // 1.5 MB
  __bf16* wprojT  = (__bf16*)(ws + off); off += (size_t)DMODEL * DMODEL * 2;     // 0.5 MB
  __bf16* q       = (__bf16*)(ws + off); off += BS * DMODEL * 2;                 // 8 MB
  __bf16* k       = (__bf16*)(ws + off); off += BS * DMODEL * 2;                 // 8 MB
  __bf16* vT      = (__bf16*)(ws + off); off += BS * DMODEL * 2;                 // 8 MB
  __bf16* attnout = (__bf16*)(ws + off); off += BS * DMODEL * 2;                 // 8 MB

  int nx = (int)(BS * DMODEL);
  k_cast_x<<<(nx + 255) / 256, 256, 0, stream>>>(x, xb, nx);
  k_transpose_cast<<<(3 * DMODEL * DMODEL + 255) / 256, 256, 0, stream>>>(
      w_qkv, wqkvT, DMODEL, 3 * DMODEL);
  k_transpose_cast<<<(DMODEL * DMODEL + 255) / 256, 256, 0, stream>>>(
      w_proj, wprojT, DMODEL, DMODEL);

  k_qkv_gemm<<<dim3(BS / 128, (3 * DMODEL) / 64), 128, 0, stream>>>(xb, wqkvT, q, k, vT);
  k_band_attn<<<BATCH * NH * (S_LEN / 64), 128, 0, stream>>>(q, k, vT, attnout);
  k_proj_gemm<<<dim3(BS / 128, DMODEL / 64), 128, 0, stream>>>(attnout, wprojT, out);
}
